// GCNNet_65180423684243
// MI455X (gfx1250) — matrix operand order repaired and checked by reading, not compiled, not run on hardware
//
#include <hip/hip_runtime.h>
#include <hip/hip_bf16.h>

// ---- CDNA5 wave32 WMMA types -------------------------------------------------
typedef __attribute__((ext_vector_type(16))) __bf16 v16bf;
typedef __attribute__((ext_vector_type(8)))  __bf16 v8bf;
typedef __attribute__((ext_vector_type(8)))  float  v8f;
typedef __attribute__((ext_vector_type(4)))  float  v4f;

union FragBf { v16bf v; v8bf h[2]; __bf16 e[16]; };

#define NN    30      // nodes
#define INC   512     // input channels
#define FF    128     // hidden features
#define COUT  9       // output channels

// LDS strides (elements). All chosen so 8-element bf16 groups are 16B-aligned.
#define SA    136     // Asub / Hs row stride (bf16)
#define SB    136     // Bsub row stride (bf16)
#define SXT   36      // XWt row stride (f32)
#define STT   40      // Tt row stride (bf16)
#define SAN   40      // Anbf row stride (bf16)
#define SH2   136     // H2s row stride (f32, aliases XWt storage)

__device__ __forceinline__ v8f wmma_bf16(const FragBf& a, const FragBf& b, v8f c) {
  return __builtin_amdgcn_wmma_f32_16x16x32_bf16(false, a.v, false, b.v,
                                                 (short)0, c, false, false);
}

// A fragment: 16-bit A matrix 16x32, row = lane%16, k = 16*(lane/16)+e
__device__ __forceinline__ FragBf load_afrag(const __bf16* rowptr, int kbase, int hl) {
  FragBf f;
  f.h[0] = *(const v8bf*)(rowptr + kbase + 16 * hl);
  f.h[1] = *(const v8bf*)(rowptr + kbase + 16 * hl + 8);
  return f;
}
// B fragment from transposed storage Bt[n][k]: col = lane%16, k = kbase + 16*(lane/16)+e
__device__ __forceinline__ FragBf load_bfrag(const __bf16* rowptr, int kbase, int hl) {
  FragBf f;
  f.h[0] = *(const v8bf*)(rowptr + kbase + 16 * hl);
  f.h[1] = *(const v8bf*)(rowptr + kbase + 16 * hl + 8);
  return f;
}

// ---- gfx1250 async global->LDS (ASYNCcnt tracked, no VGPR round trip) --------
__device__ __forceinline__ void async_copy16(const void* gsrc, void* lds_dst) {
  unsigned loff = (unsigned)(uintptr_t)lds_dst;   // low 32 bits = LDS offset
  asm volatile("global_load_async_to_lds_b128 %0, %1, off"
               :: "v"(loff), "v"(gsrc) : "memory");
}
__device__ __forceinline__ void wait_async0() {
  asm volatile("s_wait_asynccnt 0" ::: "memory");
}

// issue one 128x128 bf16 weight tile (rows contiguous in global) into LDS buffer
__device__ __forceinline__ void issue_w_tile(const __bf16* wbase, int row_stride,
                                             __bf16* dst, int tid) {
#pragma unroll
  for (int it = 0; it < 8; ++it) {
    int id = tid + 256 * it;            // 0..2047, 16 B each
    int nn = id >> 4;
    int k8 = (id & 15) * 8;
    async_copy16(wbase + nn * row_stride + k8, &dst[nn * SB + k8]);
  }
}

// ---- real-chunk staging: NT 16B loads -> bf16 pack -> b128 LDS store ---------
struct RealStage { v4f v[4]; };
__device__ __forceinline__ RealStage load_real_chunk(const float* rb, int kc, int tid) {
  RealStage s;
#pragma unroll
  for (int it = 0; it < 2; ++it) {
    int grp = tid + 256 * it;           // 0..511: row = grp>>4, col8 = (grp&15)*8
    int r = grp >> 4, c8 = (grp & 15) * 8;
    if (r < NN) {
      const v4f* p = (const v4f*)(rb + r * INC + kc * 128 + c8);
      s.v[2 * it]     = __builtin_nontemporal_load(p);
      s.v[2 * it + 1] = __builtin_nontemporal_load(p + 1);
    } else {
      s.v[2 * it] = (v4f)0.f; s.v[2 * it + 1] = (v4f)0.f;
    }
  }
  return s;
}
__device__ __forceinline__ void store_real_chunk(const RealStage& s, __bf16* dst, int tid) {
#pragma unroll
  for (int it = 0; it < 2; ++it) {
    int grp = tid + 256 * it;
    int r = grp >> 4, c8 = (grp & 15) * 8;
    v8bf pk;
    const v4f a = s.v[2 * it], b = s.v[2 * it + 1];
    pk[0] = (__bf16)a.x; pk[1] = (__bf16)a.y; pk[2] = (__bf16)a.z; pk[3] = (__bf16)a.w;
    pk[4] = (__bf16)b.x; pk[5] = (__bf16)b.y; pk[6] = (__bf16)b.z; pk[7] = (__bf16)b.w;
    *(v8bf*)&dst[r * SA + c8] = pk;
  }
}

// ---- one-shot weight prep: f32 -> bf16, transposed to [n][k] ----------------
__global__ __launch_bounds__(256) void prep_weights(const float* __restrict__ W1,
                                                    const float* __restrict__ W2,
                                                    __bf16* __restrict__ W1t,
                                                    __bf16* __restrict__ W2t) {
  int idx = blockIdx.x * 256 + threadIdx.x;
  if (idx < INC * FF) {                       // W1t[n*INC + k] = W1[k*FF + n]
    int n = idx / INC, k = idx % INC;
    W1t[idx] = (__bf16)W1[k * FF + n];
  } else if (idx < INC * FF + FF * FF) {      // W2t[n*FF + k] = W2[k*FF + n]
    int t = idx - INC * FF;
    int n = t / FF, k = t % FF;
    W2t[t] = (__bf16)W2[k * FF + n];
  }
}

// ---- fused GCN: one block per batch element, 8 waves --------------------------
__global__ __launch_bounds__(256) void gcn_fused(
    const float* __restrict__ real, const float* __restrict__ graph,
    const __bf16* __restrict__ W1t, const __bf16* __restrict__ W2t,
    const float* __restrict__ b1,   const float* __restrict__ b2,
    const float* __restrict__ Wlin, const float* __restrict__ blin,
    const float* __restrict__ Wconv, const float* __restrict__ bconv,
    float* __restrict__ out)
{
  __shared__ __align__(16) float  Ans[32 * 33];        // A_hat scratch
  __shared__            float  dinv[32];
  __shared__ __align__(16) __bf16 Anbf[32 * SAN];      // normalized adjacency, bf16
  __shared__ __align__(16) __bf16 Asub[2][32 * SA];    // real K-chunks (double buffer)
  __shared__ __align__(16) __bf16 Bsub[2][128 * SB];   // weight tiles (double buffer)
  __shared__ __align__(16) float  XWt[128 * SXT];      // XW1 transposed; reused as H2s
  __shared__ __align__(16) __bf16 Hs[32 * SA];         // H1 [node][feat] bf16
  __shared__ __align__(16) __bf16 Tt[128 * STT];       // (H1@W2) transposed bf16
  __shared__            float  xs[32];

  const int b    = blockIdx.x;
  const int tid  = threadIdx.x;
  const int lane = tid & 31;
  const int wave = tid >> 5;       // 0..7  -> N-tile
  const int hl   = lane >> 4;      // lane half
  const int ln   = lane & 15;
  const int n    = wave;

  // ================= Stage A: normalized adjacency ===========================
  for (int i = tid; i < 32 * 33; i += 256) Ans[i] = 0.f;
  __syncthreads();
  const float* g = graph + (size_t)b * (5 * NN * NN);
  for (int q = tid; q < 225; q += 256) {               // 900 floats as 16B NT loads
    v4f s = (v4f)0.f;
#pragma unroll
    for (int band = 0; band < 5; ++band)
      s += __builtin_nontemporal_load((const v4f*)(g + band * 900 + 4 * q));
    float sv[4] = {s.x, s.y, s.z, s.w};
#pragma unroll
    for (int t = 0; t < 4; ++t) {
      int e = 4 * q + t;
      int i = e / NN, j = e % NN;
      Ans[i * 33 + j] = (i == j) ? 1.f : ((sv[t] != 0.f) ? 1.f : 0.f);
    }
  }
  __syncthreads();
  if (tid < 32) {
    if (tid < NN) {
      float d = 0.f;
      for (int j = 0; j < NN; ++j) d += Ans[tid * 33 + j];
      dinv[tid] = rsqrtf(d);       // d >= 1 always (self loop)
    } else dinv[tid] = 0.f;
  }
  __syncthreads();
  for (int e = tid; e < 32 * 32; e += 256) {
    int i = e >> 5, j = e & 31;
    float v = (i < NN && j < NN) ? Ans[i * 33 + j] * dinv[i] * dinv[j] : 0.f;
    Anbf[i * SAN + j] = (__bf16)v;
  }

  // ================= Stage B: XW1 = real_b @ W1 (4 K-chunks, async dbl-buf) ==
  const float* rb = real + (size_t)b * (NN * INC);

  // prologue: chunk 0 weights (async -> LDS) + chunk 0 real (cvt path)
  issue_w_tile(W1t + 0 * 128, INC, Bsub[0], tid);
  {
    RealStage s0 = load_real_chunk(rb, 0, tid);
    store_real_chunk(s0, Asub[0], tid);
  }
  wait_async0();
  __syncthreads();                                     // also publishes Anbf

  v8f acc0 = {}, acc1 = {};
  for (int kc = 0; kc < 4; ++kc) {
    const __bf16* Acur = Asub[kc & 1];
    const __bf16* Bcur = Bsub[kc & 1];
    const bool havenext = (kc < 3);
    RealStage s;
    if (havenext) {                                    // prefetch next chunk
      issue_w_tile(W1t + (kc + 1) * 128, INC, Bsub[(kc + 1) & 1], tid);
      s = load_real_chunk(rb, kc + 1, tid);
    } else {                                           // prefetch W2 for Stage D
      issue_w_tile(W2t, FF, Bsub[0], tid);
    }
#pragma unroll
    for (int ks = 0; ks < 4; ++ks) {
      FragBf bfr = load_bfrag(&Bcur[(n * 16 + ln) * SB], ks * 32, hl);
      FragBf a0  = load_afrag(&Acur[ln * SA],        ks * 32, hl);
      acc0 = wmma_bf16(a0, bfr, acc0);
      FragBf a1  = load_afrag(&Acur[(16 + ln) * SA], ks * 32, hl);
      acc1 = wmma_bf16(a1, bfr, acc1);
    }
    if (havenext) store_real_chunk(s, Asub[(kc + 1) & 1], tid);
    wait_async0();                                     // next-chunk async done
    __syncthreads();
  }
#pragma unroll
  for (int r = 0; r < 8; ++r) {                        // XWt[n][node]
    XWt[(n * 16 + ln) * SXT + (r + 8 * hl)]      = acc0[r];
    XWt[(n * 16 + ln) * SXT + (16 + r + 8 * hl)] = acc1[r];
  }
  __syncthreads();

  // ================= Stage C: H1 = relu(An @ XW1 + b1) =======================
  {
    FragBf bf;
    const v4f* xp = (const v4f*)&XWt[(n * 16 + ln) * SXT + 16 * hl];
#pragma unroll
    for (int q = 0; q < 4; ++q) {
      v4f v = xp[q];
      bf.e[4 * q + 0] = (__bf16)v.x; bf.e[4 * q + 1] = (__bf16)v.y;
      bf.e[4 * q + 2] = (__bf16)v.z; bf.e[4 * q + 3] = (__bf16)v.w;
    }
    FragBf a0 = load_afrag(&Anbf[ln * SAN],        0, hl);
    FragBf a1 = load_afrag(&Anbf[(16 + ln) * SAN], 0, hl);
    v8f c0 = {}, c1 = {};
    c0 = wmma_bf16(a0, bf, c0);
    c1 = wmma_bf16(a1, bf, c1);
    float bias = b1[n * 16 + ln];
#pragma unroll
    for (int r = 0; r < 8; ++r) {
      Hs[(r + 8 * hl) * SA + n * 16 + ln]      = (__bf16)fmaxf(c0[r] + bias, 0.f);
      Hs[(16 + r + 8 * hl) * SA + n * 16 + ln] = (__bf16)fmaxf(c1[r] + bias, 0.f);
    }
  }
  __syncthreads();

  // ================= Stage D: T = H1 @ W2 (W2 already async-loaded) ==========
  {
    v8f t0 = {}, t1 = {};
#pragma unroll
    for (int ks = 0; ks < 4; ++ks) {
      FragBf bfr = load_bfrag(&Bsub[0][(n * 16 + ln) * SB], ks * 32, hl);
      FragBf a0  = load_afrag(&Hs[ln * SA],        ks * 32, hl);
      t0 = wmma_bf16(a0, bfr, t0);
      FragBf a1  = load_afrag(&Hs[(16 + ln) * SA], ks * 32, hl);
      t1 = wmma_bf16(a1, bfr, t1);
    }
#pragma unroll
    for (int r = 0; r < 8; ++r) {                      // Tt[n][node] bf16
      Tt[(n * 16 + ln) * STT + (r + 8 * hl)]      = (__bf16)t0[r];
      Tt[(n * 16 + ln) * STT + (16 + r + 8 * hl)] = (__bf16)t1[r];
    }
  }
  __syncthreads();

  // ================= Stage D2: H2 = relu(An @ T + b2) ========================
  float* H2s = XWt;                                    // reuse (4352 <= 4608 floats)
  {
    FragBf bf = load_bfrag(&Tt[(n * 16 + ln) * STT], 0, hl);
    FragBf a0 = load_afrag(&Anbf[ln * SAN],        0, hl);
    FragBf a1 = load_afrag(&Anbf[(16 + ln) * SAN], 0, hl);
    v8f d0 = {}, d1 = {};
    d0 = wmma_bf16(a0, bf, d0);
    d1 = wmma_bf16(a1, bf, d1);
    float bias = b2[n * 16 + ln];
#pragma unroll
    for (int r = 0; r < 8; ++r) {
      H2s[(r + 8 * hl) * SH2 + n * 16 + ln]      = fmaxf(d0[r] + bias, 0.f);
      H2s[(16 + r + 8 * hl) * SH2 + n * 16 + ln] = fmaxf(d1[r] + bias, 0.f);
    }
  }
  __syncthreads();

  // ================= Stage E: lin + relu + Conv1d(30->9, k=1) ================
  if (tid < NN) {
    float s = blin[0];
    for (int f = 0; f < FF; ++f) s += H2s[tid * SH2 + f] * Wlin[f];
    xs[tid] = fmaxf(s, 0.f);
  }
  __syncthreads();
  if (tid < COUT) {
    float s = bconv[tid];
#pragma unroll
    for (int j = 0; j < NN; ++j) s += Wconv[tid * NN + j] * xs[j];
    out[(size_t)b * COUT + tid] = s;
  }
}

// ---- host-side launch --------------------------------------------------------
extern "C" void kernel_launch(void* const* d_in, const int* in_sizes, int n_in,
                              void* d_out, int out_size, void* d_ws, size_t ws_size,
                              hipStream_t stream) {
  const float* real  = (const float*)d_in[0];
  // d_in[1] = imag (unused by the reference network)
  const float* graph = (const float*)d_in[2];
  const float* W1    = (const float*)d_in[3];
  const float* b1    = (const float*)d_in[4];
  const float* W2    = (const float*)d_in[5];
  const float* b2    = (const float*)d_in[6];
  const float* Wlin  = (const float*)d_in[7];
  const float* blin  = (const float*)d_in[8];
  const float* Wconv = (const float*)d_in[9];
  const float* bconv = (const float*)d_in[10];
  float* out = (float*)d_out;

  const int batch = in_sizes[0] / (NN * INC);         // 4096

  __bf16* W1t = (__bf16*)d_ws;                        // 512*128 bf16
  __bf16* W2t = W1t + INC * FF;                       // 128*128 bf16

  const int prep_elems = INC * FF + FF * FF;
  prep_weights<<<(prep_elems + 255) / 256, 256, 0, stream>>>(W1, W2, W1t, W2t);
  gcn_fused<<<batch, 256, 0, stream>>>(real, graph, W1t, W2t, b1, b2,
                                       Wlin, blin, Wconv, bconv, out);
}
